// AdaptiveSparseReservoir_1245540516172
// MI455X (gfx1250) — matrix operand order repaired and checked by reading, not compiled, not run on hardware
//
#include <hip/hip_runtime.h>

#define D_IN  4096
#define UNITS 4096
#define NNZ   262144
#define BATCH 1024

typedef __attribute__((ext_vector_type(16))) _Float16 v16h;
typedef __attribute__((ext_vector_type(8)))  _Float16 v8h;
typedef __attribute__((ext_vector_type(4)))  _Float16 v4h;
typedef __attribute__((ext_vector_type(8)))  float    v8f;
typedef __attribute__((ext_vector_type(4)))  float    v4f;
typedef __attribute__((ext_vector_type(4)))  unsigned int u32x4;
typedef __attribute__((ext_vector_type(8)))  unsigned int u32x8;

// ---------------- Phase 1: zero the f16 hi/lo weight planes (64 MB) ---------
__global__ void asr_zero(float* __restrict__ p) {
  size_t i = (size_t)blockIdx.x * blockDim.x + threadIdx.x;
  v4f z = {0.f, 0.f, 0.f, 0.f};
  ((v4f*)p)[i] = z;
}

// 16-bit atomic add via 32-bit CAS (collisions are ~2k of 16.7M entries)
__device__ inline void atomic_add_f16(_Float16* addr, _Float16 val) {
  size_t a = (size_t)addr;
  unsigned int* base = (unsigned int*)(a & ~(size_t)3);
  const unsigned sh = (a & 2) ? 16u : 0u;
  unsigned int cur = *base;
  while (true) {
    unsigned short hb = (unsigned short)(cur >> sh);
    _Float16 sum = __builtin_bit_cast(_Float16, hb) + val;
    unsigned short sb = __builtin_bit_cast(unsigned short, sum);
    unsigned int nv = (cur & ~(0xFFFFu << sh)) | ((unsigned int)sb << sh);
    unsigned int prev = atomicCAS(base, cur, nv);
    if (prev == cur) break;
    cur = prev;
  }
}

// ---------------- Phase 2: scatter COO values into Wh/Wl f16 planes ---------
__global__ void asr_scatter(const float* __restrict__ values,
                            const int*  __restrict__ rows,
                            const int*  __restrict__ cols,
                            _Float16* __restrict__ Wh,
                            _Float16* __restrict__ Wl) {
  int k = blockIdx.x * blockDim.x + threadIdx.x;
  if (k < NNZ) {
    float v = values[k];
    _Float16 h = (_Float16)v;
    _Float16 l = (_Float16)(v - (float)h);
    size_t idx = (size_t)cols[k] * D_IN + rows[k];
    atomic_add_f16(&Wh[idx], h);
    atomic_add_f16(&Wl[idx], l);
  }
}

// ---------------- Phase 3: split X into hi/lo f16 planes --------------------
__global__ void asr_split(const float* __restrict__ src,
                          _Float16* __restrict__ hi,
                          _Float16* __restrict__ lo) {
  size_t i = ((size_t)blockIdx.x * blockDim.x + threadIdx.x) * 4;
  v4f f = *(const v4f*)(src + i);
  v4h h, l;
  #pragma unroll
  for (int q = 0; q < 4; ++q) {
    _Float16 hv = (_Float16)f[q];
    h[q] = hv;
    l[q] = (_Float16)(f[q] - (float)hv);
  }
  *(v4h*)(hi + i) = h;
  *(v4h*)(lo + i) = l;
}

// ---------------- Phase 4: dense GEMM, TDM-staged split-f16 WMMA ------------
// out[b][n] = relu( sum_k X[b][k] * Wt[n][k] + bias[n] )
// C += Ah*Bh + Ah*Bl + Al*Bh  (fp32 accumulate, ~2^-22 relative error)
__launch_bounds__(256)
__global__ void asr_gemm(const _Float16* __restrict__ Xh,  // [BATCH][D_IN]
                         const _Float16* __restrict__ Xl,
                         const _Float16* __restrict__ Wh,  // [UNITS][D_IN]
                         const _Float16* __restrict__ Wl,
                         const float* __restrict__ bias,
                         float* __restrict__ out) {
  // [buf][slab: 0=Ah 1=Al 2=Bh 3=Bl][row][k]  row-major, 8 KB per slab, 64 KB total
  __shared__ __align__(64) _Float16 sm[2][4][128][32];

  const int tid  = threadIdx.x;
  const int lane = tid & 31;
  const int wave = tid >> 5;
  const int wm   = wave >> 2;   // 0..1 -> 64 rows of C per wave
  const int wn   = wave & 3;    // 0..3 -> 32 cols of C per wave

  const int mb = blockIdx.y * 128;
  const int nb = blockIdx.x * 128;

  // Issue 4 TDM descriptors: 2D tile 32 halves x 128 rows, row stride D_IN.
  auto issue = [&](int buf, int k0) {
    const _Float16* gsrc[4] = {
      Xh + (size_t)mb * D_IN + k0, Xl + (size_t)mb * D_IN + k0,
      Wh + (size_t)nb * D_IN + k0, Wl + (size_t)nb * D_IN + k0 };
    const unsigned tdim1[4] = { BATCH, BATCH, UNITS, UNITS };
    #pragma unroll
    for (int s = 0; s < 4; ++s) {
      unsigned long long ga = (unsigned long long)gsrc[s];
      unsigned int lds = (unsigned int)(size_t)(&sm[buf][s][0][0]);
      u32x4 g0;
      g0[0] = 1u;                                   // count=1, user mode
      g0[1] = lds;                                  // lds_addr
      g0[2] = (unsigned int)ga;                     // global_addr[31:0]
      g0[3] = (unsigned int)((ga >> 32) & 0x1FFFFFFu) | (2u << 30); // [56:32] | type=2
      u32x8 g1;
      g1[0] = 1u << 16;                             // wg_mask=0 | data_size=1 (2B)
      g1[1] = ((unsigned)D_IN) << 16;               // tensor_dim0[15:0]
      g1[2] = (tdim1[s] & 0xFFFFu) << 16;           // dim0[31:16]=0 | tensor_dim1[15:0]
      g1[3] = 32u << 16;                            // dim1[31:16]=0 | tile_dim0=32
      g1[4] = 128u;                                 // tile_dim1=128 | tile_dim2=0
      g1[5] = (unsigned)D_IN;                       // tensor_dim0_stride[31:0]
      g1[6] = 0u;                                   // stride0 hi | stride1 lo
      g1[7] = 0u;                                   // stride1 hi
      u32x4 gz;
      gz[0] = 0u; gz[1] = 0u; gz[2] = 0u; gz[3] = 0u;
      asm volatile("tensor_load_to_lds %0, %1, %2, %3"
                   :: "s"(g0), "s"(g1), "s"(gz), "s"(gz)
                   : "memory");
    }
  };

  v8f c[4][2] = {};

  if (wave == 0) {
    issue(0, 0);
    __builtin_amdgcn_s_wait_tensorcnt(0);
  }
  __syncthreads();

  const int KT = D_IN / 32;
  for (int kt = 0; kt < KT; ++kt) {
    const int cur = kt & 1;
    if ((kt + 1 < KT) && wave == 0) issue(cur ^ 1, (kt + 1) * 32);

    // B operands: lane<16 -> K 0-15, lane>=16 -> K 16-31 (contiguous 32 B)
    v16h bh[2], bl[2];
    #pragma unroll
    for (int j = 0; j < 2; ++j) {
      const int rowB = (wn * 2 + j) * 16 + (lane & 15);
      const int kB = (lane < 16) ? 0 : 16;
      bh[j] = *(const v16h*)&sm[cur][2][rowB][kB];
      bl[j] = *(const v16h*)&sm[cur][3][rowB][kB];
    }
    #pragma unroll
    for (int i = 0; i < 4; ++i) {
      // A operands: lane<16 -> K 0-15, lane>=16 -> K 16-31
      const int rowA = (wm * 4 + i) * 16 + (lane & 15);
      const int kA = (lane < 16) ? 0 : 16;
      v8h ah0 = *(const v8h*)&sm[cur][0][rowA][kA];
      v8h ah1 = *(const v8h*)&sm[cur][0][rowA][kA + 8];
      v8h al0 = *(const v8h*)&sm[cur][1][rowA][kA];
      v8h al1 = *(const v8h*)&sm[cur][1][rowA][kA + 8];
      v16h ah = __builtin_shufflevector(ah0, ah1, 0,1,2,3,4,5,6,7,8,9,10,11,12,13,14,15);
      v16h al = __builtin_shufflevector(al0, al1, 0,1,2,3,4,5,6,7,8,9,10,11,12,13,14,15);
      #pragma unroll
      for (int j = 0; j < 2; ++j) {
        c[i][j] = __builtin_amdgcn_wmma_f32_16x16x32_f16(
            false, ah, false, bh[j], (short)0, c[i][j], false, false);
        c[i][j] = __builtin_amdgcn_wmma_f32_16x16x32_f16(
            false, ah, false, bl[j], (short)0, c[i][j], false, false);
        c[i][j] = __builtin_amdgcn_wmma_f32_16x16x32_f16(
            false, al, false, bh[j], (short)0, c[i][j], false, false);
      }
    }
    if (wave == 0) __builtin_amdgcn_s_wait_tensorcnt(0);
    __syncthreads();
  }

  // Epilogue: bias + ReLU. C layout: lane<16 -> N=lane, M=q; lane>=16 -> N=lane-16, M=q+8
  #pragma unroll
  for (int i = 0; i < 4; ++i) {
    const int m0 = mb + (wm * 4 + i) * 16 + ((lane >> 4) << 3);
    #pragma unroll
    for (int j = 0; j < 2; ++j) {
      const int n = nb + (wn * 2 + j) * 16 + (lane & 15);
      const float bv = bias[n];
      #pragma unroll
      for (int q = 0; q < 8; ++q) {
        float v = c[i][j][q] + bv;
        out[(size_t)(m0 + q) * UNITS + n] = v > 0.f ? v : 0.f;
      }
    }
  }
}

extern "C" void kernel_launch(void* const* d_in, const int* in_sizes, int n_in,
                              void* d_out, int out_size, void* d_ws, size_t ws_size,
                              hipStream_t stream) {
  (void)in_sizes; (void)n_in; (void)out_size; (void)ws_size;
  const float* x      = (const float*)d_in[0];
  const float* values = (const float*)d_in[1];
  const float* bias   = (const float*)d_in[2];
  const int*   rows   = (const int*)d_in[3];
  const int*   cols   = (const int*)d_in[4];
  float* out = (float*)d_out;

  // workspace layout (f16 planes): Wh 32MB | Wl 32MB | Xh 8MB | Xl 8MB
  _Float16* Wh = (_Float16*)d_ws;
  _Float16* Wl = Wh + (size_t)UNITS * D_IN;
  _Float16* Xh = Wl + (size_t)UNITS * D_IN;
  _Float16* Xl = Xh + (size_t)BATCH * D_IN;

  // zero Wh||Wl (contiguous 64 MB = 4M float4)
  asr_zero<<<(2 * (size_t)UNITS * D_IN * 2) / (16 * 256), 256, 0, stream>>>((float*)Wh);
  asr_scatter<<<NNZ / 256, 256, 0, stream>>>(values, rows, cols, Wh, Wl);
  asr_split<<<((size_t)BATCH * D_IN) / (4 * 256), 256, 0, stream>>>(x, Xh, Xl);

  dim3 grid(UNITS / 128, BATCH / 128);
  asr_gemm<<<grid, 256, 0, stream>>>(Xh, Xl, Wh, Wl, bias, out);
}
